// LAAT_47416438947914
// MI455X (gfx1250) — compile-verified, matrix operand order checked by reading, not run on hardware
//
#include <hip/hip_runtime.h>
#include <hip/hip_bf16.h>
#include <math.h>
#include <stdint.h>

// ---------------- problem constants ----------------
#define BDIM 8
#define SEQ  2048
#define EDIM 300
#define EPAD 320      // 300 padded to multiple of 32 for WMMA K
#define HDIM 512
#define G4   2048     // 4*HDIM
#define RNN  1024     // 2*HDIM
#define DA   512
#define NCLS 8921

typedef __attribute__((ext_vector_type(16))) __bf16  v16bf;
typedef __attribute__((ext_vector_type(8)))  float   v8f;
typedef __attribute__((ext_vector_type(4)))  unsigned v4u;
typedef __attribute__((ext_vector_type(8)))  int     v8i;
typedef __attribute__((ext_vector_type(4)))  int     v4i;

#if defined(__HIP_DEVICE_COMPILE__) && __has_builtin(__builtin_amdgcn_tensor_load_to_lds) && __has_builtin(__builtin_amdgcn_s_wait_tensorcnt)
#define USE_TDM 1
#else
#define USE_TDM 0
#endif

// ---------------- small device helpers ----------------
__device__ inline v8f wmma_bf16(v16bf a, v16bf b, v8f c) {
#if defined(__HIP_DEVICE_COMPILE__)
  return __builtin_amdgcn_wmma_f32_16x16x32_bf16(false, a, false, b, (short)0, c, false, false);
#else
  return c;
#endif
}
__device__ inline void tensor_wait0() {
#if USE_TDM
  __builtin_amdgcn_s_wait_tensorcnt(0);
#endif
}
__device__ inline void nap() {
#if defined(__HIP_DEVICE_COMPILE__)
  __builtin_amdgcn_s_sleep(1);
#endif
}

// Load a 16x32 bf16 fragment from LDS following the documented wave32 layout:
// lanes 0-15: rows 0..15, K 0..7 & 16..23 ; lanes 16-31: rows 0..15, K 8..15 & 24..31
__device__ inline v16bf ldfrag(const __bf16* p, int row0, int ld, int lane) {
  int r  = row0 + (lane & 15);
  int k0 = (lane >> 4) << 3;
  struct P { uint4 a, b; } t;
  t.a = *(const uint4*)(p + (size_t)r * ld + k0);
  t.b = *(const uint4*)(p + (size_t)r * ld + k0 + 16);
  return __builtin_bit_cast(v16bf, t);
}

// TDM: DMA a [128 rows x 32 elems] bf16 tile from global (row stride ld_elems)
// into LDS at lds_byte_off.  OOB rows/cols (vs m_rem/k_rem) read zero.
// 6-arg builtin form: (g0 u32x4, g1 i32x8, g2 i32x4, g3 i32x4, i32x8, cpol)
__device__ inline void tdm_load_tile(unsigned lds_byte_off, const void* gptr,
                                     unsigned k_rem, unsigned m_rem, unsigned ld_elems) {
#if USE_TDM
  unsigned long long ga = (unsigned long long)(size_t)gptr;
  v4u g0 = {0, 0, 0, 0};
  g0.x = 1u;                                   // count=1, user descriptor
  g0.y = lds_byte_off;                         // lds_addr (bytes)
  g0.z = (unsigned)ga;                         // global_addr[31:0]
  g0.w = (unsigned)((ga >> 32) & 0x01FFFFFFu) | (2u << 30);  // addr[56:32] | type=2
  v8i g1 = {0, 0, 0, 0, 0, 0, 0, 0};
  g1[0] = (int)(1u << 16);                     // wg_mask=0, data_size=1 (2 bytes)
  g1[1] = (int)((k_rem & 0xFFFFu) << 16);      // tensor_dim0[15:0]
  g1[2] = (int)(((k_rem >> 16) & 0xFFFFu) | ((m_rem & 0xFFFFu) << 16));  // dim0 hi | dim1 lo
  g1[3] = (int)(((m_rem >> 16) & 0xFFFFu) | (32u << 16));                // dim1 hi | tile_dim0=32
  g1[4] = (int)128;                            // tile_dim1=128, tile_dim2=0
  g1[5] = (int)ld_elems;                       // tensor_dim0_stride lo32
  v4i z4 = {0, 0, 0, 0};
  v8i z8 = {0, 0, 0, 0, 0, 0, 0, 0};
  __builtin_amdgcn_tensor_load_to_lds(g0, g1, z4, z4, z8, 0);
#else
  (void)lds_byte_off; (void)gptr; (void)k_rem; (void)m_rem; (void)ld_elems;
#endif
}

// Fallback: manual 128x32 bf16 tile stage (256 threads)
__device__ inline void stage_bf16_manual(__bf16* dst, const __bf16* src,
                                         int row0, int rows_total, int k0, int ld, int tid) {
  int r  = tid >> 1;
  int c0 = (tid & 1) << 4;
  uint4 q0 = {0, 0, 0, 0}, q1 = {0, 0, 0, 0};
  int gr = row0 + r;
  if (gr < rows_total) {
    const uint4* s = (const uint4*)(src + (size_t)gr * ld + k0 + c0);
    q0 = s[0]; q1 = s[1];
  }
  uint4* d = (uint4*)(dst + r * 32 + c0);
  d[0] = q0; d[1] = q1;
}

__device__ inline float wave_max(float v) {
  for (int o = 16; o; o >>= 1) v = fmaxf(v, __shfl_xor(v, o, 32));
  return v;
}
__device__ inline float wave_sum(float v) {
  for (int o = 16; o; o >>= 1) v += __shfl_xor(v, o, 32);
  return v;
}
__device__ inline float sigf(float x) { return 1.f / (1.f + __expf(-x)); }

// ---------------- prep kernels ----------------
__global__ __launch_bounds__(256) void init_state_kernel(__bf16* hbuf, unsigned* cnt) {
  int i = blockIdx.x * 256 + threadIdx.x;
  if (i < 2 * 16 * HDIM) hbuf[i] = (__bf16)0.f;
  if (i < 2) cnt[i] = 0u;
}

// fp32 [rows, sk] -> bf16 [rows, dk], zero-padded columns
__global__ __launch_bounds__(256) void cvt_pad_kernel(const float* __restrict__ src,
                                                      __bf16* __restrict__ dst,
                                                      int rows, int sk, int dk) {
  long long idx = (long long)blockIdx.x * 256 + threadIdx.x;
  long long total = (long long)rows * dk;
  if (idx >= total) return;
  int k = (int)(idx % dk);
  long long r = idx / dk;
  float v = (k < sk) ? src[r * sk + k] : 0.f;
  dst[idx] = (__bf16)v;
}

// X[s*B + b][EPAD] = embed[text[b][s]][:] (zero-padded 300->320), bf16
__global__ __launch_bounds__(256) void embed_gather_kernel(const int* __restrict__ text,
                                                           const float* __restrict__ emb,
                                                           __bf16* __restrict__ X) {
  long long idx = (long long)blockIdx.x * 256 + threadIdx.x;
  const long long total = (long long)SEQ * BDIM * EPAD;
  if (idx >= total) return;
  int e = (int)(idx % EPAD);
  long long m = idx / EPAD;
  int b = (int)(m % BDIM);
  int s = (int)(m / BDIM);
  float v = 0.f;
  if (e < EDIM) {
    int tok = text[(size_t)b * SEQ + s];
    v = emb[(size_t)tok * EDIM + e];
  }
  X[idx] = (__bf16)v;
}

// ---------------- generic WMMA GEMM: Out = act(A[MxK] * W[NxK]^T (+bias)) ----------------
// 128x128 tile, BK=32, 256 threads (8 waves in 4x2), bf16 WMMA, fp32 accum.
template <bool A_F32, bool TANH, bool OUT_BF16, bool TRANS, bool BIAS>
__global__ __launch_bounds__(256) void wmma_gemm_kernel(
    const void* __restrict__ Ap, int lda, long long a_bstride,
    const __bf16* __restrict__ Wp, int ldw, long long w_bstride,
    const float* __restrict__ bias,
    void* __restrict__ Outp, int ldo, long long o_bstride,
    int M, int N, int K, int trans_S, long long trans_cs) {
  __shared__ __bf16 sA[2][128 * 32];
  __shared__ __bf16 sB[2][128 * 32];
  const int tid = threadIdx.x, lane = tid & 31, wv = tid >> 5;
  const int wm = wv & 3, wn = wv >> 2;      // wave grid 4 (M) x 2 (N)
  const int m0 = blockIdx.x * 128, n0 = blockIdx.y * 128;
  const int bz = blockIdx.z;
  const __bf16* A16 = (const __bf16*)Ap;
  const float*  A32 = (const float*)Ap;
  if (A_F32) A32 += (long long)bz * a_bstride; else A16 += (long long)bz * a_bstride;
  const __bf16* W = Wp + (long long)bz * w_bstride;
  const int ksteps = K >> 5;

  auto stageA = [&](int k, int buf) {
    const int k0 = k << 5;
    if (A_F32) {  // load fp32, convert to bf16 while staging to LDS
      int r = tid >> 1, c0 = (tid & 1) << 4;
      union { __bf16 h[16]; uint4 q[2]; } u;
      int gm = m0 + r;
      if (gm < M) {
        const float* s = A32 + (size_t)gm * lda + k0 + c0;
#pragma unroll
        for (int i = 0; i < 16; ++i) u.h[i] = (__bf16)s[i];
      } else {
#pragma unroll
        for (int i = 0; i < 16; ++i) u.h[i] = (__bf16)0.f;
      }
      uint4* d = (uint4*)(&sA[buf][r * 32 + c0]);
      d[0] = u.q[0]; d[1] = u.q[1];
    } else {
#if USE_TDM
      if (wv == 0)
        tdm_load_tile((unsigned)(size_t)&sA[buf][0], A16 + (size_t)m0 * lda + k0,
                      (unsigned)(K - k0), (unsigned)(M - m0), (unsigned)lda);
#else
      stage_bf16_manual(&sA[buf][0], A16, m0, M, k0, lda, tid);
#endif
    }
  };
  auto stageB = [&](int k, int buf) {
    const int k0 = k << 5;
#if USE_TDM
    if (wv == 0)
      tdm_load_tile((unsigned)(size_t)&sB[buf][0], W + (size_t)n0 * ldw + k0,
                    (unsigned)(K - k0), (unsigned)(N - n0), (unsigned)ldw);
#else
    stage_bf16_manual(&sB[buf][0], W, n0, N, k0, ldw, tid);
#endif
  };

  v8f acc[2][4] = {};
  stageA(0, 0); stageB(0, 0);
  for (int k = 0; k < ksteps; ++k) {
#if USE_TDM
    if (wv == 0) tensor_wait0();      // tile k complete (TENSORcnt in-order)
#endif
    __syncthreads();                  // tile k visible; buffer (k+1)&1 free
    if (k + 1 < ksteps) { stageA(k + 1, (k + 1) & 1); stageB(k + 1, (k + 1) & 1); }
    const __bf16* a = sA[k & 1];
    const __bf16* bb = sB[k & 1];
    v16bf af[2], bf_[4];
#pragma unroll
    for (int i = 0; i < 2; ++i) af[i] = ldfrag(a, wm * 32 + i * 16, 32, lane);
#pragma unroll
    for (int j = 0; j < 4; ++j) bf_[j] = ldfrag(bb, wn * 64 + j * 16, 32, lane);
#pragma unroll
    for (int i = 0; i < 2; ++i)
#pragma unroll
      for (int j = 0; j < 4; ++j) acc[i][j] = wmma_bf16(af[i], bf_[j], acc[i][j]);
    __syncthreads();
  }

  // epilogue: C layout VGPR v -> M = v (+8 for upper half-wave), N = lane&15
#pragma unroll
  for (int i = 0; i < 2; ++i)
#pragma unroll
    for (int j = 0; j < 4; ++j)
#pragma unroll
      for (int v = 0; v < 8; ++v) {
        int m = m0 + wm * 32 + i * 16 + v + ((lane >> 4) << 3);
        int n = n0 + wn * 64 + j * 16 + (lane & 15);
        if (m < M && n < N) {
          float val = acc[i][j][v];
          if (BIAS) val += bias[n];
          if (TANH) val = tanhf(val);
          if (TRANS) {
            long long b = m / trans_S;
            int s = m % trans_S;
            ((float*)Outp)[b * trans_cs + (long long)n * trans_S + s] = val;
          } else {
            size_t o = (size_t)bz * o_bstride + (size_t)m * ldo + n;
            if (OUT_BF16) ((__bf16*)Outp)[o] = (__bf16)val;
            else          ((float*)Outp)[o] = val;
          }
        }
      }
}

// ---------------- persistent bidirectional LSTM scan ----------------
// grid = 16 WGs: dir = blockIdx.x>>3, slice = blockIdx.x&7.  Each WG owns 64 hidden
// units (256 gate columns), keeps its Whh slice resident in LDS (256KB of 320KB),
// and runs the full 2048-step scan with a device-scope arrive/spin barrier per step.
__global__ __launch_bounds__(256) void lstm_scan_kernel(
    const float* __restrict__ xw,      // [2][S][B][4H] fp32 (bias already folded)
    const __bf16* __restrict__ Whh,    // [2][4H][H] bf16
    __bf16* __restrict__ hbuf,         // [2][16][H] bf16 (rows 8..15 stay zero)
    unsigned* __restrict__ cnt,        // [2] arrive counters
    __bf16* __restrict__ Hbf,          // [B][S][RNN]
    __bf16* __restrict__ HTbf) {       // [B][RNN][S]
  extern __shared__ char smem[];
  __bf16* whh = (__bf16*)smem;                                 // [256][512]
  __bf16* hsh = (__bf16*)(smem + 256 * HDIM * 2);              // [16][512]
  float*  gsh = (float*)(smem + 256 * HDIM * 2 + 16 * HDIM * 2);  // [16][256]
  float*  csh = gsh + 16 * 256;                                // [8][64]

  const int tid = threadIdx.x, lane = tid & 31, wv = tid >> 5;
  const int dir = blockIdx.x >> 3, slice = blockIdx.x & 7;
  const __bf16* whh_g = Whh + (size_t)dir * G4 * HDIM;

  // load this WG's 256 gate rows (gathered: gate g, hidden j in [slice*64, slice*64+64))
  for (int i = tid; i < 256 * (HDIM / 8); i += 256) {
    int n = i >> 6, kq = i & 63;
    int gate = n >> 6, jj = n & 63;
    int grow = gate * HDIM + slice * 64 + jj;
    *(uint4*)(whh + (size_t)n * HDIM + kq * 8) =
        *(const uint4*)(whh_g + (size_t)grow * HDIM + kq * 8);
  }
  for (int i = tid; i < 512; i += 256) csh[i] = 0.f;
  __syncthreads();

  for (int t = 0; t < SEQ; ++t) {
    const int s = dir ? (SEQ - 1 - t) : t;
    // bring full h (16x512, b rows 8..15 are zero pad) into LDS
    for (int i = tid; i < 16 * HDIM / 8; i += 256)
      ((uint4*)hsh)[i] = ((const uint4*)(hbuf + (size_t)dir * 16 * HDIM))[i];
    __syncthreads();

    // g[b, n] = h @ Whh_slice^T : M=16(pad), N=32 per wave, K=512
    v8f acc0 = {}, acc1 = {};
#pragma unroll 4
    for (int kk = 0; kk < HDIM / 32; ++kk) {
      v16bf a  = ldfrag(hsh, 0, HDIM, lane);
      v16bf b0 = ldfrag(whh, wv * 32, HDIM, lane);
      v16bf b1 = ldfrag(whh, wv * 32 + 16, HDIM, lane);
      acc0 = wmma_bf16(a, b0, acc0);
      acc1 = wmma_bf16(a, b1, acc1);
      hsh += 32; whh += 32;   // advance K
    }
    hsh -= HDIM; whh -= HDIM;
#pragma unroll
    for (int v = 0; v < 8; ++v) {
      int mrow = v + ((lane >> 4) << 3);
      gsh[mrow * 256 + wv * 32 + (lane & 15)]      = acc0[v];
      gsh[mrow * 256 + wv * 32 + 16 + (lane & 15)] = acc1[v];
    }
    __syncthreads();

    // gate nonlinearities + state update for (b, jj)
    for (int it = tid; it < 512; it += 256) {
      int b = it >> 6, jj = it & 63;
      int jg = slice * 64 + jj;
      const float* xr = xw + (((size_t)dir * SEQ + s) * BDIM + b) * G4;
      float gi = gsh[b * 256 + 0   + jj] + xr[0 * HDIM + jg];
      float gf = gsh[b * 256 + 64  + jj] + xr[1 * HDIM + jg];
      float gg = gsh[b * 256 + 128 + jj] + xr[2 * HDIM + jg];
      float go = gsh[b * 256 + 192 + jj] + xr[3 * HDIM + jg];
      float c = sigf(gf) * csh[it] + sigf(gi) * tanhf(gg);
      csh[it] = c;
      float h = sigf(go) * tanhf(c);
      __bf16 hb = (__bf16)h;
      hbuf[(size_t)dir * 16 * HDIM + (size_t)b * HDIM + jg] = hb;
      Hbf[((size_t)b * SEQ + s) * RNN + dir * HDIM + jg] = hb;
      HTbf[((size_t)b * RNN + dir * HDIM + jg) * SEQ + s] = hb;
    }
    __threadfence();
    __syncthreads();
    if (tid == 0) {   // device-scope step barrier across the 8 WGs of this direction
      atomicAdd(&cnt[dir], 1u);
      unsigned target = 8u * (unsigned)(t + 1);
      while (__hip_atomic_load(&cnt[dir], __ATOMIC_ACQUIRE, __HIP_MEMORY_SCOPE_AGENT) < target)
        nap();
    }
    __syncthreads();
  }
}

// ---------------- softmax over S, in place on [B*C, S] fp32 ----------------
__global__ __launch_bounds__(256) void softmax_rows_kernel(float* __restrict__ alpha,
                                                           long long rows) {
  long long row = (long long)blockIdx.x * 8 + (threadIdx.x >> 5);
  if (row >= rows) return;
  float* p = alpha + row * SEQ;
  int lane = threadIdx.x & 31;
  float m = -INFINITY;
  for (int i = lane; i < SEQ; i += 32) m = fmaxf(m, p[i]);
  m = wave_max(m);
  float sum = 0.f;
  for (int i = lane; i < SEQ; i += 32) sum += __expf(p[i] - m);
  sum = wave_sum(sum);
  float inv = 1.f / sum;
  for (int i = lane; i < SEQ; i += 32) p[i] = __expf(p[i] - m) * inv;
}

// ---------------- logits[b,c] = Ow[c,:] . E[b,c,:] + Ob[c] ----------------
__global__ __launch_bounds__(256) void logits_kernel(const float* __restrict__ Ow,
                                                     const float* __restrict__ Ob,
                                                     const __bf16* __restrict__ E,
                                                     float* __restrict__ out,
                                                     long long rows) {
  long long row = (long long)blockIdx.x * 8 + (threadIdx.x >> 5);
  if (row >= rows) return;
  int lane = threadIdx.x & 31;
  int c = (int)(row % NCLS);
  const __bf16* e = E + (size_t)row * RNN;
  const float* w = Ow + (size_t)c * RNN;
  float s = 0.f;
  for (int i = lane; i < RNN; i += 32) s += w[i] * (float)e[i];
  s = wave_sum(s);
  if (lane == 0) out[row] = s + Ob[c];
}

// ---------------- host orchestration ----------------
extern "C" void kernel_launch(void* const* d_in, const int* in_sizes, int n_in,
                              void* d_out, int out_size, void* d_ws, size_t ws_size,
                              hipStream_t stream) {
  (void)in_sizes; (void)n_in; (void)out_size; (void)ws_size;
  const int*   text  = (const int*)d_in[0];
  // d_in[1] = length (full-length sequences; packing is identity) -- unused
  const float* emb   = (const float*)d_in[2];
  const float* Wih_f = (const float*)d_in[3];
  const float* Whh_f = (const float*)d_in[4];
  const float* b_f   = (const float*)d_in[5];
  const float* Wih_b = (const float*)d_in[6];
  const float* Whh_b = (const float*)d_in[7];
  const float* b_b   = (const float*)d_in[8];
  const float* Ww    = (const float*)d_in[9];
  const float* Qw    = (const float*)d_in[10];
  const float* Ow    = (const float*)d_in[11];
  const float* Ob    = (const float*)d_in[12];

  // workspace carve-out
  size_t off = 0;
  auto wsa = [&](size_t bytes) -> void* {
    void* p = (char*)d_ws + off;
    off += (bytes + 255) & ~(size_t)255;
    return p;
  };
  __bf16* Xbf    = (__bf16*)wsa((size_t)SEQ * BDIM * EPAD * 2);
  __bf16* WihFbf = (__bf16*)wsa((size_t)G4 * EPAD * 2);
  __bf16* WihBbf = (__bf16*)wsa((size_t)G4 * EPAD * 2);
  __bf16* Whhbf  = (__bf16*)wsa((size_t)2 * G4 * HDIM * 2);   // [dir][4H][H]
  __bf16* Wwbf   = (__bf16*)wsa((size_t)DA * RNN * 2);
  __bf16* Qwbf   = (__bf16*)wsa((size_t)NCLS * DA * 2);
  float*  xw     = (float*)wsa((size_t)2 * SEQ * BDIM * G4 * 4);
  __bf16* Hbf    = (__bf16*)wsa((size_t)BDIM * SEQ * RNN * 2);
  __bf16* HTbf   = (__bf16*)wsa((size_t)BDIM * RNN * SEQ * 2);
  __bf16* Zbf    = (__bf16*)wsa((size_t)SEQ * BDIM * DA * 2);
  __bf16* Ebf    = (__bf16*)wsa((size_t)BDIM * NCLS * RNN * 2);
  __bf16* hbuf   = (__bf16*)wsa((size_t)2 * 16 * HDIM * 2);
  unsigned* cnt  = (unsigned*)wsa(256);

  float* logits = (float*)d_out;
  float* alpha  = (float*)d_out + (size_t)BDIM * NCLS;   // [B,C,S] fp32

  // 0) reset recurrent state + step counters (must happen every call)
  init_state_kernel<<<64, 256, 0, stream>>>(hbuf, cnt);

  // 1) weights -> bf16 (pad Wih K 300->320)
  auto cvt = [&](const float* s, __bf16* d, int rows, int sk, int dk) {
    long long tot = (long long)rows * dk;
    cvt_pad_kernel<<<(unsigned)((tot + 255) / 256), 256, 0, stream>>>(s, d, rows, sk, dk);
  };
  cvt(Wih_f, WihFbf, G4, EDIM, EPAD);
  cvt(Wih_b, WihBbf, G4, EDIM, EPAD);
  cvt(Whh_f, Whhbf, G4, HDIM, HDIM);
  cvt(Whh_b, Whhbf + (size_t)G4 * HDIM, G4, HDIM, HDIM);
  cvt(Ww, Wwbf, DA, RNN, RNN);
  cvt(Qw, Qwbf, NCLS, DA, DA);

  // 2) embedding gather (bf16, padded)
  {
    long long tot = (long long)SEQ * BDIM * EPAD;
    embed_gather_kernel<<<(unsigned)((tot + 255) / 256), 256, 0, stream>>>(text, emb, Xbf);
  }

  // 3) input projections xw = X @ Wih^T + b   (M=16384, N=2048, K=320)
  wmma_gemm_kernel<false, false, false, false, true>
      <<<dim3(128, 16, 1), 256, 0, stream>>>(
          Xbf, EPAD, 0, WihFbf, EPAD, 0, b_f, xw, G4, 0,
          SEQ * BDIM, G4, EPAD, 0, 0);
  wmma_gemm_kernel<false, false, false, false, true>
      <<<dim3(128, 16, 1), 256, 0, stream>>>(
          Xbf, EPAD, 0, WihBbf, EPAD, 0, b_b,
          xw + (size_t)SEQ * BDIM * G4, G4, 0,
          SEQ * BDIM, G4, EPAD, 0, 0);

  // 4) bidirectional LSTM scan (16 persistent WGs, ~290KB dynamic LDS each)
  {
    size_t smem = (size_t)256 * HDIM * 2 + 16 * HDIM * 2 + 16 * 256 * 4 + 8 * 64 * 4;
    lstm_scan_kernel<<<16, 256, smem, stream>>>(xw, Whhbf, hbuf, cnt, Hbf, HTbf);
  }

  // 5) Z = tanh(H @ Ww^T) -> bf16   (M=16384, N=512, K=1024)
  wmma_gemm_kernel<false, true, true, false, false>
      <<<dim3(128, 4, 1), 256, 0, stream>>>(
          Hbf, RNN, 0, Wwbf, RNN, 0, nullptr, Zbf, DA, 0,
          SEQ * BDIM, DA, RNN, 0, 0);

  // 6) scores = Z @ Qw^T, stored transposed as alpha[b, c, s] fp32 in d_out
  wmma_gemm_kernel<false, false, false, true, false>
      <<<dim3(128, (NCLS + 127) / 128, 1), 256, 0, stream>>>(
          Zbf, DA, 0, Qwbf, DA, 0, nullptr, alpha, 0, 0,
          SEQ * BDIM, NCLS, DA, SEQ, (long long)NCLS * SEQ);

  // 7) softmax over S, in place (wave per (b,c) row)
  softmax_rows_kernel<<<(unsigned)(((long long)BDIM * NCLS + 7) / 8), 256, 0, stream>>>(
      alpha, (long long)BDIM * NCLS);

  // 8) E[b] = alpha[b] @ H[b]   (batched over b: M=8921, N=1024, K=2048; A fp32->bf16 on stage)
  wmma_gemm_kernel<true, false, true, false, false>
      <<<dim3((NCLS + 127) / 128, RNN / 128, BDIM), 256, 0, stream>>>(
          alpha, SEQ, (long long)NCLS * SEQ,
          HTbf, SEQ, (long long)RNN * SEQ, nullptr,
          Ebf, RNN, (long long)NCLS * RNN,
          NCLS, RNN, SEQ, 0, 0);

  // 9) logits
  logits_kernel<<<(unsigned)(((long long)BDIM * NCLS + 7) / 8), 256, 0, stream>>>(
      Ow, Ob, Ebf, logits, (long long)BDIM * NCLS);
}
